// CopyMB_62758062129367
// MI455X (gfx1250) — matrix operand order repaired and checked by reading, not compiled, not run on hardware
//
#include <hip/hip_runtime.h>
#include <hip/hip_bf16.h>

typedef _Float16 half_t;
typedef __attribute__((ext_vector_type(16))) _Float16 v16h;
typedef __attribute__((ext_vector_type(8)))  float    v8f;

#define BB    8
#define LL    128
#define EE    256
#define HH    256
#define BIOd  64
#define TTc   3
#define VRR   51
#define NSTEP 11
#define BLL   (BB*LL)
#define CATD  (HH+BIOd)   // 320

__device__ __forceinline__ float sigf(float x){ return 1.0f/(1.0f+expf(-x)); }

// ---------------------------------------------------------------------------
// Generic WMMA GEMM:  C[M,N] = A[M,K] * W[N,K]^T + bias
// A, W row-major f16; C f32 (optional); Ch f16 (optional, optionally relu'd).
// block = 128 threads (4 waves); one 16x16 tile per wave.
// grid = (N/64, M/16); requires N%64==0, M%16==0, K%32==0.
// ---------------------------------------------------------------------------
__global__ void k_gemm(const half_t* __restrict__ A, const half_t* __restrict__ W,
                       const float* __restrict__ bias, float* __restrict__ C,
                       half_t* __restrict__ Ch, int M, int N, int K, int reluH)
{
  const int tid  = threadIdx.x;
  const int wave = tid >> 5;
  const int lane = tid & 31;
  const int grp  = lane >> 4;
  const int lr   = lane & 15;
  const int tileN = blockIdx.x*4 + wave;
  const int tileM = blockIdx.y;

  const half_t* Arow = A + (size_t)(tileM*16 + lr)*K;
  const half_t* Wrow = W + (size_t)(tileN*16 + lr)*K;

  v8f acc = {0,0,0,0,0,0,0,0};
  for (int k0 = 0; k0 < K; k0 += 32) {
    v16h a, b;
#pragma unroll
    for (int v = 0; v < 8; ++v) {
      int kb = ((v>>2)<<4) + (grp<<3) + ((v&3)<<1);
      a[2*v]   = Arow[k0+kb];
      a[2*v+1] = Arow[k0+kb+1];
    }
#pragma unroll
    for (int e = 0; e < 16; ++e) b[e] = Wrow[k0 + ((e>>3)<<4) + (grp<<3) + (e&7)];
    acc = __builtin_amdgcn_wmma_f32_16x16x32_f16(false, a, false, b, (short)0, acc, false, false);
  }
  const int n = tileN*16 + lr;
  const float bv = bias ? bias[n] : 0.0f;
#pragma unroll
  for (int r = 0; r < 8; ++r) {
    int m = tileM*16 + r + grp*8;
    float v = acc[r] + bv;
    if (C)  C[(size_t)m*N + n] = v;
    if (Ch) Ch[(size_t)m*N + n] = (half_t)(reluH ? (v > 0.0f ? v : 0.0f) : v);
  }
}

// ---------------------------------------------------------------------------
// Sequential GRU encoder: one workgroup per direction (blockIdx.x = dir).
// 256 threads = 8 waves. h lives in LDS (f16 padded to 16 rows for WMMA A).
// gi (precomputed x@W_ih^T + b_ih) read from global, per timestep.
// ---------------------------------------------------------------------------
__global__ void k_enc_gru(const half_t* __restrict__ WhhF, const half_t* __restrict__ WhhB,
                          const float* __restrict__ bhhF,  const float* __restrict__ bhhB,
                          const float* __restrict__ giF,   const float* __restrict__ giB,
                          float* __restrict__ oF,          float* __restrict__ oB)
{
  const int dir = blockIdx.x;
  const half_t* Whh = dir ? WhhB : WhhF;
  const float*  bhh = dir ? bhhB : bhhF;
  const float*  gi  = dir ? giB  : giF;
  float*        o   = dir ? oB   : oF;

  __shared__ half_t h16[16*HH];      // 8 KB, rows 8..15 stay zero (pad)
  __shared__ float  gh[BB*3*HH];     // 24 KB
  __shared__ float  h32[BB*HH];      // 8 KB

  const int tid = threadIdx.x;
  for (int i = tid; i < 16*HH; i += 256) h16[i] = (half_t)0.0f;
  for (int i = tid; i < BB*HH; i += 256) h32[i] = 0.0f;
  __syncthreads();

  const int wave = tid >> 5;
  const int lane = tid & 31;
  const int grp  = lane >> 4;
  const int lr   = lane & 15;

  for (int t = 0; t < LL; ++t) {
    const int tt = dir ? (LL-1-t) : t;
    // gh = h @ Whh^T + bhh  (48 N-tiles, 6 per wave)
    for (int nt = wave*6; nt < wave*6+6; ++nt) {
      v8f acc = {0,0,0,0,0,0,0,0};
      const half_t* Arow = h16 + lr*HH;
      const half_t* Wrow = Whh + (size_t)(nt*16+lr)*HH;
      for (int k0 = 0; k0 < HH; k0 += 32) {
        v16h a, b;
#pragma unroll
        for (int v = 0; v < 8; ++v) {
          int kb = ((v>>2)<<4) + (grp<<3) + ((v&3)<<1);
          a[2*v]   = Arow[k0+kb];
          a[2*v+1] = Arow[k0+kb+1];
        }
#pragma unroll
        for (int e = 0; e < 16; ++e) b[e] = Wrow[k0 + ((e>>3)<<4) + (grp<<3) + (e&7)];
        acc = __builtin_amdgcn_wmma_f32_16x16x32_f16(false, a, false, b, (short)0, acc, false, false);
      }
      int n = nt*16 + lr;
      float bv = bhh[n];
#pragma unroll
      for (int r = 0; r < 8; ++r) {
        int m = r + grp*8;
        if (m < BB) gh[m*(3*HH) + n] = acc[r] + bv;
      }
    }
    __syncthreads();
    // gates + state update
    for (int idx = tid; idx < BB*HH; idx += 256) {
      int b = idx >> 8, c = idx & 255;
      size_t row = (size_t)(b*LL + tt);
      float ir = gi[row*768 + c], iz = gi[row*768 + 256 + c], inn = gi[row*768 + 512 + c];
      float hr = gh[b*768 + c],   hz = gh[b*768 + 256 + c],   hn  = gh[b*768 + 512 + c];
      float r = sigf(ir + hr);
      float z = sigf(iz + hz);
      float n = tanhf(inn + r*hn);
      float hnew = (1.0f - z)*n + z*h32[idx];
      h32[idx] = hnew;
      h16[b*HH + c] = (half_t)hnew;
      o[row*HH + c] = hnew;
    }
    __syncthreads();
  }
}

// ------------------------- small helper kernels ----------------------------
__global__ void k_cvt_f16(const float* __restrict__ src, int srcLd, int srcRows,
                          int cols, int dstRows, half_t* __restrict__ dst)
{
  int idx = blockIdx.x*256 + threadIdx.x;
  if (idx >= dstRows*cols) return;
  int r = idx / cols, c = idx % cols;
  float v = (r < srcRows) ? src[(size_t)r*srcLd + c] : 0.0f;
  dst[idx] = (half_t)v;
}

__global__ void k_padf32(const float* __restrict__ src, int n, int dstN, float* __restrict__ dst)
{
  int idx = blockIdx.x*256 + threadIdx.x;
  if (idx >= dstN) return;
  dst[idx] = (idx < n) ? src[idx] : 0.0f;
}

__global__ void k_embed(const int* __restrict__ tokens, const float* __restrict__ word_emb,
                        half_t* __restrict__ emb_h)
{
  int idx = blockIdx.x*256 + threadIdx.x;
  if (idx >= BLL*EE) return;
  int j = idx >> 8, c = idx & 255;
  emb_h[idx] = (half_t)word_emb[(size_t)tokens[j]*EE + c];
}

__global__ void k_combine_o(const float* __restrict__ of, const float* __restrict__ ob,
                            float* __restrict__ o)
{
  int idx = blockIdx.x*256 + threadIdx.x;
  if (idx >= BLL*HH) return;
  o[idx] = 0.5f*(of[idx] + ob[idx]);
}

__global__ void k_emi(const float* __restrict__ o, const float* __restrict__ emi_w,
                      const float* __restrict__ emi_b, float* __restrict__ emi)
{
  int idx = blockIdx.x*256 + threadIdx.x;
  if (idx >= BLL*TTc) return;
  int j = idx / TTc, t = idx % TTc;
  const float* orow = o + (size_t)j*HH;
  const float* w = emi_w + (size_t)t*HH;
  float s = emi_b[t];
  for (int k = 0; k < HH; ++k) s += orow[k]*w[k];
  emi[idx] = s;
}

__global__ void k_crf(const float* __restrict__ emi, const int* __restrict__ tokens,
                      const int* __restrict__ tags, const float* __restrict__ start,
                      const float* __restrict__ endv, const float* __restrict__ trans,
                      float* __restrict__ crf_out)
{
  __shared__ float lsh[BB];
  int tid = threadIdx.x;
  if (tid < BB) {
    int b = tid;
    const int* tok = tokens + b*LL;
    const int* tg  = tags + b*LL;
    const float* em = emi + (size_t)b*LL*TTc;
    float score = start[tg[0]] + em[tg[0]];
    int msum = (tok[0] != 0) ? 1 : 0;
    for (int t = 1; t < LL; ++t) {
      float m = (tok[t] != 0) ? 1.0f : 0.0f;
      msum += (tok[t] != 0) ? 1 : 0;
      score += (trans[tg[t-1]*TTc + tg[t]] + em[t*TTc + tg[t]]) * m;
    }
    int last = msum - 1; if (last < 0) last = 0;
    score += endv[tg[last]];
    float al[TTc];
    for (int c = 0; c < TTc; ++c) al[c] = start[c] + em[c];
    for (int t = 1; t < LL; ++t) {
      if (tok[t] != 0) {
        float na[TTc];
        for (int c = 0; c < TTc; ++c) {
          float mx = -1e30f;
          for (int p = 0; p < TTc; ++p) mx = fmaxf(mx, al[p] + trans[p*TTc+c]);
          float s = 0.0f;
          for (int p = 0; p < TTc; ++p) s += expf(al[p] + trans[p*TTc+c] - mx);
          na[c] = logf(s) + mx + em[t*TTc + c];
        }
        for (int c = 0; c < TTc; ++c) al[c] = na[c];
      }
    }
    float mx = -1e30f;
    for (int c = 0; c < TTc; ++c) mx = fmaxf(mx, al[c] + endv[c]);
    float s = 0.0f;
    for (int c = 0; c < TTc; ++c) s += expf(al[c] + endv[c] - mx);
    lsh[b] = -(score - (logf(s) + mx));
  }
  __syncthreads();
  if (tid == 0) {
    float s = 0.0f;
    for (int b = 0; b < BB; ++b) s += lsh[b];
    crf_out[0] = s / (float)BB;
  }
}

__global__ void k_cat_rc(const float* __restrict__ o, const int* __restrict__ bio_gold,
                         const float* __restrict__ bio_emb, half_t* __restrict__ rc)
{
  int idx = blockIdx.x*256 + threadIdx.x;
  if (idx >= BLL*CATD) return;
  int j = idx / CATD, c = idx % CATD;
  float v = (c < HH) ? o[(size_t)j*HH + c] : bio_emb[bio_gold[j]*BIOd + (c-HH)];
  rc[idx] = (half_t)(v > 0.0f ? v : 0.0f);
}

__global__ void k_init_hd(const float* __restrict__ o, const int* __restrict__ lengths,
                          float* __restrict__ hd, half_t* __restrict__ hd_h,
                          half_t* __restrict__ hdrelu_h)
{
  int idx = blockIdx.x*256 + threadIdx.x;
  if (idx >= BLL*HH) return;
  int j = idx >> 8, c = idx & 255;
  int b = j / LL;
  int t = lengths[b] - 1; if (t < 0) t = 0;
  float v = o[((size_t)(b*LL + t))*HH + c];
  hd[idx] = v;
  hd_h[idx] = (half_t)v;
  hdrelu_h[idx] = (half_t)(v > 0.0f ? v : 0.0f);
}

__global__ void k_gru_elem(const float* __restrict__ gi, const float* __restrict__ gh,
                           const float* __restrict__ hprev, float* __restrict__ hout,
                           half_t* __restrict__ h_h, half_t* __restrict__ hrelu_h)
{
  int idx = blockIdx.x*256 + threadIdx.x;
  if (idx >= BLL*HH) return;
  int j = idx >> 8, c = idx & 255;
  size_t row = (size_t)j*768;
  float r = sigf(gi[row+c] + gh[row+c]);
  float z = sigf(gi[row+256+c] + gh[row+256+c]);
  float n = tanhf(gi[row+512+c] + r*gh[row+512+c]);
  float hnew = (1.0f - z)*n + z*hprev[idx];
  hout[idx] = hnew;
  h_h[idx] = (half_t)hnew;
  hrelu_h[idx] = (half_t)(hnew > 0.0f ? hnew : 0.0f);
}

// odd-step logits: logits[j,l] = b2 + sum_h w2[h]*relu(U[j,h] + V[(j%B)*L+l, h])
__global__ void k_pair_logits(const float* __restrict__ U, const float* __restrict__ V,
                              const float* __restrict__ w2, const float* __restrict__ b2,
                              float* __restrict__ logits)
{
  int j = blockIdx.x;
  int tid = threadIdx.x;   // 128 threads = l
  __shared__ float Us[HH], Ws[HH];
  for (int c = tid; c < HH; c += 128) { Us[c] = U[(size_t)j*HH + c]; Ws[c] = w2[c]; }
  __syncthreads();
  const float* Vp = V + ((size_t)((j & (BB-1))*LL + tid))*HH;
  float acc = 0.0f;
  for (int h = 0; h < HH; ++h) {
    float x = Us[h] + Vp[h];
    acc += Ws[h]*(x > 0.0f ? x : 0.0f);
  }
  logits[(size_t)j*LL + tid] = acc + b2[0];
}

__global__ void k_ce(const float* __restrict__ logits, int ld, int ncols,
                     const int* __restrict__ seq_gold, int step,
                     const int* __restrict__ mdec, float* __restrict__ ce_out)
{
  int j = blockIdx.x;
  const float* Lg = logits + (size_t)j*ld;
  __shared__ float red[128];
  int tid = threadIdx.x;
  float mx = -1e30f;
  for (int c = tid; c < ncols; c += 128) mx = fmaxf(mx, Lg[c]);
  red[tid] = mx; __syncthreads();
  for (int off = 64; off > 0; off >>= 1) { if (tid < off) red[tid] = fmaxf(red[tid], red[tid+off]); __syncthreads(); }
  float gmax = red[0]; __syncthreads();
  float s = 0.0f;
  for (int c = tid; c < ncols; c += 128) s += expf(Lg[c] - gmax);
  red[tid] = s; __syncthreads();
  for (int off = 64; off > 0; off >>= 1) { if (tid < off) red[tid] += red[tid+off]; __syncthreads(); }
  if (tid == 0) {
    int tgt = seq_gold[j*NSTEP + step];
    float ce = (logf(red[0]) + gmax) - Lg[tgt];
    float m = (mdec[j*NSTEP + step] != 0) ? 1.0f : 0.0f;
    ce_out[j] = ce * m;
  }
}

__global__ void k_gather_rel(const int* __restrict__ seq_gold, int step,
                             const float* __restrict__ rel_emb, half_t* __restrict__ dec_in_h)
{
  int idx = blockIdx.x*256 + threadIdx.x;
  if (idx >= BLL*HH) return;
  int j = idx >> 8, c = idx & 255;
  int tgt = seq_gold[j*NSTEP + step];
  dec_in_h[idx] = (half_t)rel_emb[(size_t)tgt*EE + c];
}

__global__ void k_gather_rc(const int* __restrict__ seq_gold, int step,
                            const half_t* __restrict__ rc, half_t* __restrict__ Xg)
{
  int idx = blockIdx.x*256 + threadIdx.x;
  if (idx >= BLL*CATD) return;
  int j = idx / CATD, c = idx % CATD;
  int tgt = seq_gold[j*NSTEP + step];
  Xg[idx] = rc[((size_t)((j & (BB-1))*LL + tgt))*CATD + c];
}

__global__ void k_zero1(float* p){ p[0] = 0.0f; }

__global__ void k_sum_accum(const float* __restrict__ v, int n, float* __restrict__ acc)
{
  __shared__ float s[256];
  int tid = threadIdx.x;
  float x = 0.0f;
  for (int i = tid; i < n; i += 256) x += v[i];
  s[tid] = x; __syncthreads();
  for (int off = 128; off > 0; off >>= 1) { if (tid < off) s[tid] += s[tid+off]; __syncthreads(); }
  if (tid == 0) acc[0] += s[0];
}

__global__ void k_msum(const int* __restrict__ m, int n, float* __restrict__ out)
{
  __shared__ float s[256];
  int tid = threadIdx.x;
  float x = 0.0f;
  for (int i = tid; i < n; i += 256) x += (m[i] != 0) ? 1.0f : 0.0f;
  s[tid] = x; __syncthreads();
  for (int off = 128; off > 0; off >>= 1) { if (tid < off) s[tid] += s[tid+off]; __syncthreads(); }
  if (tid == 0) out[0] = s[0];
}

__global__ void k_final(const float* crf, const float* total, const float* msum, float* out)
{
  out[0] = crf[0] + total[0]/msum[0];
}

// ---------------------------------------------------------------------------
static inline void gemm(const half_t* A, const half_t* W, const float* bias,
                        float* C, half_t* Ch, int M, int N, int K, int reluH,
                        hipStream_t s)
{
  dim3 grid(N/64, M/16);
  k_gemm<<<grid, 128, 0, s>>>(A, W, bias, C, Ch, M, N, K, reluH);
}

extern "C" void kernel_launch(void* const* d_in, const int* in_sizes, int n_in,
                              void* d_out, int out_size, void* d_ws, size_t ws_size,
                              hipStream_t stream)
{
  (void)in_sizes; (void)n_in; (void)out_size; (void)ws_size;
  const int*   tokens      = (const int*)d_in[0];
  const int*   bio_gold    = (const int*)d_in[1];
  const int*   seq_gold    = (const int*)d_in[2];
  const int*   mask_decode = (const int*)d_in[3];
  const int*   lengths     = (const int*)d_in[4];
  const float* word_emb    = (const float*)d_in[5];
  const float* rel_emb     = (const float*)d_in[6];
  const float* bio_emb     = (const float*)d_in[7];
  const float* enc_w_ih_f  = (const float*)d_in[8];
  const float* enc_w_hh_f  = (const float*)d_in[9];
  const float* enc_b_ih_f  = (const float*)d_in[10];
  const float* enc_b_hh_f  = (const float*)d_in[11];
  const float* enc_w_ih_b  = (const float*)d_in[12];
  const float* enc_w_hh_b  = (const float*)d_in[13];
  const float* enc_b_ih_b  = (const float*)d_in[14];
  const float* enc_b_hh_b  = (const float*)d_in[15];
  const float* dec_w_ih    = (const float*)d_in[16];
  const float* dec_w_hh    = (const float*)d_in[17];
  const float* dec_b_ih    = (const float*)d_in[18];
  const float* dec_b_hh    = (const float*)d_in[19];
  const float* crf_start   = (const float*)d_in[20];
  const float* crf_end     = (const float*)d_in[21];
  const float* crf_trans   = (const float*)d_in[22];
  const float* emi_w       = (const float*)d_in[23];
  const float* emi_b       = (const float*)d_in[24];
  const float* relb_w      = (const float*)d_in[25];
  const float* relb_b      = (const float*)d_in[26];
  const float* ent1_w      = (const float*)d_in[27];
  const float* ent1_b      = (const float*)d_in[28];
  const float* ent2_w      = (const float*)d_in[29];
  const float* ent2_b      = (const float*)d_in[30];
  const float* cat_w       = (const float*)d_in[31];
  const float* cat_b       = (const float*)d_in[32];

  char* wp = (char*)d_ws;
  auto carve = [&](size_t bytes) -> void* {
    void* r = (void*)wp;
    wp += (bytes + 255) & ~(size_t)255;
    return r;
  };

  half_t* WihF   = (half_t*)carve((size_t)768*256*2);
  half_t* WhhF   = (half_t*)carve((size_t)768*256*2);
  half_t* WihB   = (half_t*)carve((size_t)768*256*2);
  half_t* WhhB   = (half_t*)carve((size_t)768*256*2);
  half_t* Wdih   = (half_t*)carve((size_t)768*256*2);
  half_t* Wdhh   = (half_t*)carve((size_t)768*256*2);
  half_t* relbW  = (half_t*)carve((size_t)64*256*2);
  half_t* catW   = (half_t*)carve((size_t)256*CATD*2);
  half_t* W1a    = (half_t*)carve((size_t)256*256*2);
  half_t* W1b    = (half_t*)carve((size_t)256*CATD*2);
  float*  relbBp = (float*)carve((size_t)64*4);
  half_t* emb_h  = (half_t*)carve((size_t)BLL*EE*2);
  float*  gi_f   = (float*)carve((size_t)BLL*768*4);
  float*  gi_b   = (float*)carve((size_t)BLL*768*4);
  float*  of     = (float*)carve((size_t)BLL*HH*4);
  float*  ob     = (float*)carve((size_t)BLL*HH*4);
  float*  oo     = (float*)carve((size_t)BLL*HH*4);
  float*  emi    = (float*)carve((size_t)BLL*TTc*4);
  half_t* rc_h   = (half_t*)carve((size_t)BLL*CATD*2);
  float*  Vbuf   = (float*)carve((size_t)BLL*HH*4);
  float*  Ubuf   = (float*)carve((size_t)BLL*HH*4);
  half_t* dec_in = (half_t*)carve((size_t)BLL*HH*2);
  float*  hd     = (float*)carve((size_t)BLL*HH*4);
  half_t* hd_h   = (half_t*)carve((size_t)BLL*HH*2);
  half_t* hdre_h = (half_t*)carve((size_t)BLL*HH*2);
  float*  gi_d   = (float*)carve((size_t)BLL*768*4);
  float*  gh_d   = (float*)carve((size_t)BLL*768*4);
  half_t* Xg     = (half_t*)carve((size_t)BLL*CATD*2);
  float*  logits = (float*)carve((size_t)BLL*LL*4);
  float*  ce_buf = (float*)carve((size_t)BLL*4);
  float*  crfL   = (float*)carve(4);
  float*  total  = (float*)carve(4);
  float*  msum   = (float*)carve(4);

  const int BS = 256;
  auto nb = [](int n){ return (n + 255)/256; };

  // ---- weight conversions to f16 ----
  k_cvt_f16<<<nb(768*256),BS,0,stream>>>(enc_w_ih_f, 256, 768, 256, 768, WihF);
  k_cvt_f16<<<nb(768*256),BS,0,stream>>>(enc_w_hh_f, 256, 768, 256, 768, WhhF);
  k_cvt_f16<<<nb(768*256),BS,0,stream>>>(enc_w_ih_b, 256, 768, 256, 768, WihB);
  k_cvt_f16<<<nb(768*256),BS,0,stream>>>(enc_w_hh_b, 256, 768, 256, 768, WhhB);
  k_cvt_f16<<<nb(768*256),BS,0,stream>>>(dec_w_ih,   256, 768, 256, 768, Wdih);
  k_cvt_f16<<<nb(768*256),BS,0,stream>>>(dec_w_hh,   256, 768, 256, 768, Wdhh);
  k_cvt_f16<<<nb(64*256), BS,0,stream>>>(relb_w,     256, VRR, 256, 64,  relbW);
  k_cvt_f16<<<nb(256*CATD),BS,0,stream>>>(cat_w,     CATD, 256, CATD, 256, catW);
  k_cvt_f16<<<nb(256*256),BS,0,stream>>>(ent1_w,     576, 256, 256, 256, W1a);
  k_cvt_f16<<<nb(256*CATD),BS,0,stream>>>(ent1_w+HH, 576, 256, CATD, 256, W1b);
  k_padf32<<<1,64,0,stream>>>(relb_b, VRR, 64, relbBp);

  // ---- encoder ----
  k_embed<<<nb(BLL*EE),BS,0,stream>>>(tokens, word_emb, emb_h);
  gemm(emb_h, WihF, enc_b_ih_f, gi_f, nullptr, BLL, 768, EE, 0, stream);
  gemm(emb_h, WihB, enc_b_ih_b, gi_b, nullptr, BLL, 768, EE, 0, stream);
  k_enc_gru<<<2,256,0,stream>>>(WhhF, WhhB, enc_b_hh_f, enc_b_hh_b, gi_f, gi_b, of, ob);
  k_combine_o<<<nb(BLL*HH),BS,0,stream>>>(of, ob, oo);

  // ---- CRF ----
  k_emi<<<nb(BLL*TTc),BS,0,stream>>>(oo, emi_w, emi_b, emi);
  k_crf<<<1,32,0,stream>>>(emi, tokens, bio_gold, crf_start, crf_end, crf_trans, crfL);

  // ---- decoder setup ----
  k_cat_rc<<<nb(BLL*CATD),BS,0,stream>>>(oo, bio_gold, bio_emb, rc_h);
  gemm(rc_h, W1b, ent1_b, Vbuf, nullptr, BLL, HH, CATD, 0, stream);   // V (incl. ent1_b), fixed
  gemm(rc_h, catW, cat_b, nullptr, dec_in, BLL, HH, CATD, 0, stream); // dec_in0 (f16)
  k_init_hd<<<nb(BLL*HH),BS,0,stream>>>(oo, lengths, hd, hd_h, hdre_h);
  k_zero1<<<1,1,0,stream>>>(total);
  k_msum<<<1,256,0,stream>>>(mask_decode, BB*LL*NSTEP, msum);

  // ---- decoder loop ----
  for (int i = 0; i < NSTEP; ++i) {
    gemm(dec_in, Wdih, dec_b_ih, gi_d, nullptr, BLL, 768, HH, 0, stream);
    gemm(hd_h,   Wdhh, dec_b_hh, gh_d, nullptr, BLL, 768, HH, 0, stream);
    k_gru_elem<<<nb(BLL*HH),BS,0,stream>>>(gi_d, gh_d, hd, hd, hd_h, hdre_h);
    if ((i & 1) == 0) {
      gemm(hd_h, relbW, relbBp, logits, nullptr, BLL, 64, HH, 0, stream);
      k_ce<<<BLL,128,0,stream>>>(logits, 64, VRR, seq_gold, i, mask_decode, ce_buf);
      k_sum_accum<<<1,256,0,stream>>>(ce_buf, BLL, total);
      k_gather_rel<<<nb(BLL*HH),BS,0,stream>>>(seq_gold, i, rel_emb, dec_in);
    } else {
      gemm(hdre_h, W1a, nullptr, Ubuf, nullptr, BLL, HH, HH, 0, stream);
      k_pair_logits<<<BLL,128,0,stream>>>(Ubuf, Vbuf, ent2_w, ent2_b, logits);
      k_ce<<<BLL,128,0,stream>>>(logits, LL, LL, seq_gold, i, mask_decode, ce_buf);
      k_sum_accum<<<1,256,0,stream>>>(ce_buf, BLL, total);
      k_gather_rc<<<nb(BLL*CATD),BS,0,stream>>>(seq_gold, i, rc_h, Xg);
      gemm(Xg, catW, cat_b, nullptr, dec_in, BLL, HH, CATD, 0, stream);
    }
  }

  k_final<<<1,1,0,stream>>>(crfL, total, msum, (float*)d_out);
}
